// MixerLayerKAN_20401094656062
// MI455X (gfx1250) — matrix operand order repaired and checked by reading, not compiled, not run on hardware
//
#include <hip/hip_runtime.h>
#include <hip/hip_bf16.h>

// ---------------------------------------------------------------------------
// KAN-Mixer layer for MI455X (gfx1250), wave32 + WMMA f16 (f32 accumulate).
// B=32, T=512, C=512, TD=256, G=3.  All four GEMMs run through
// v_wmma_f32_16x16x32_f16.
//  - Fourier features generated on the fly, branchless: sin(x)=cos(x-pi/2).
//  - B matrices stored N-major (O x K) by the prep kernels so both LDS tiles
//    are staged with ds_store_b128 (no in-LDS transpose).
//  - Block tile 128x128, 8 waves (4M x 2N), per-wave 32x64 = 2x4 fragments
//    -> 8 v_wmma per wave per 32-wide K-step (64 per block per step).
// ---------------------------------------------------------------------------

typedef __attribute__((ext_vector_type(16))) _Float16 v16h;
typedef __attribute__((ext_vector_type(8)))  _Float16 v8h;
typedef __attribute__((ext_vector_type(8)))  float    v8f;
typedef __attribute__((ext_vector_type(4)))  float    v4f;

#define BB 32
#define TT 512
#define CC 512
#define TD 256
#define GG 3
#define NROWS (BB * TT)      // = B*C = 16384 as well

__device__ __forceinline__ v8f zero8() {
    v8f z;
    #pragma unroll
    for (int i = 0; i < 8; ++i) z[i] = 0.0f;
    return z;
}

union V16U { v16h v; v8h h[2]; };

// ---------------------------------------------------------------------------
// LayerNorm over last dim (C=512).  One block per (b,t) row, 256 threads.
// transpose!=0: write dst[(b*C + c)*T + t]  (token-mixing A rows)
// transpose==0: write dst[row*C + c]
// ---------------------------------------------------------------------------
__global__ __launch_bounds__(256)
void ln_kernel(const float* __restrict__ src, const float* __restrict__ w,
               const float* __restrict__ b, float* __restrict__ dst,
               int transpose) {
    const unsigned row = blockIdx.x;            // row in [0, B*T)
    const unsigned tid = threadIdx.x;
    const float* xr = src + row * (unsigned)CC;
    float v0 = xr[tid];
    float v1 = xr[tid + 256];

    __shared__ float s0[256], s1[256];
    s0[tid] = v0 + v1;
    s1[tid] = v0 * v0 + v1 * v1;
    __syncthreads();
    #pragma unroll
    for (int off = 128; off > 0; off >>= 1) {
        if (tid < (unsigned)off) { s0[tid] += s0[tid + off]; s1[tid] += s1[tid + off]; }
        __syncthreads();
    }
    const float mean = s0[0] * (1.0f / (float)CC);
    const float var  = s1[0] * (1.0f / (float)CC) - mean * mean;
    const float rs   = rsqrtf(var + 1e-5f);

    const unsigned bb = row >> 9;  // T == 512
    const unsigned t  = row & 511;
    #pragma unroll
    for (int j = 0; j < 2; ++j) {
        const unsigned c = tid + j * 256;
        const float v = (j == 0) ? v0 : v1;
        const float y = (v - mean) * rs * w[c] + b[c];
        if (transpose)
            dst[(bb << 18) + (c << 9) + t] = y;   // [b][c][t]
        else
            dst[row * (unsigned)CC + c] = y;
    }
}

// ---------------------------------------------------------------------------
// Repack coef (2, O, 512, 3) f32 -> W (O, 3072) f16, N-major:
//   W[o][k] with k = g*512 + t; g<3 -> cos block (coef[0], mult g+1),
//   g>=3 -> sin block (coef[1], mult g-2).
// Grid: (3072/256, O); coalesced writes along k.
// ---------------------------------------------------------------------------
__global__ __launch_bounds__(256)
void prep_w_kernel(const float* __restrict__ coef, _Float16* __restrict__ W,
                   int O) {
    const unsigned k = blockIdx.x * 256 + threadIdx.x;   // 0..3071
    const unsigned o = blockIdx.y;
    const unsigned g = k >> 9;
    const unsigned t = k & 511;
    const unsigned s  = (g >= 3) ? 1u : 0u;
    const unsigned gg = g - s * 3;
    const float c = coef[((s * O + o) * 512 + t) * GG + gg];
    W[o * 3072u + k] = (_Float16)c;
}

// lw (R, Cc) f32 -> f16 copy (already N-major: rows = output col, cols = K)
__global__ __launch_bounds__(256)
void cvt_f16_kernel(const float* __restrict__ src, _Float16* __restrict__ dst) {
    const unsigned idx = blockIdx.x * 256 + threadIdx.x;
    dst[idx] = (_Float16)src[idx];
}

// ---------------------------------------------------------------------------
// WMMA GEMM:  out[n, col] = sum_k A[n,k] * Bw[col, k] (+ bias[col]) (+resid)
//   Bw is N-major (Ntotal x K row-major).
//   FEAT: A[n, k] generated on the fly from f32 src (n, 512):
//         k = g*512 + t,  feature = cos(mult*v + phase), phase=-pi/2 for sin.
//   OUT_F16: store f16 (KAN output, feeds next GEMM as A).
//   else   : f32 out with residual add; TRANS_OUT un-transposes (b,c,t)->(b,t,c).
// ---------------------------------------------------------------------------
#define LDA 40   // f16 stride (80B, 16B-aligned rows)
#define LDB 40

template<bool FEAT, bool OUT_F16, bool TRANS_OUT>
__global__ __launch_bounds__(256)
void wmma_gemm_kernel(const void* __restrict__ Ap,
                      const _Float16* __restrict__ Bw,
                      const float* __restrict__ bias,
                      const float* __restrict__ resid,
                      void* __restrict__ outp,
                      int K, int N) {
    __shared__ __align__(16) _Float16 lA[128 * LDA];   // 128 rows x 32 K
    __shared__ __align__(16) _Float16 lB[128 * LDB];   // 128 cols x 32 K

    const int tid  = threadIdx.x;
    const int lane = tid & 31;
    const int l16  = lane & 15;
    const int half = lane >> 4;
    const int wv   = tid >> 5;        // 0..7
    const int mw   = wv >> 1;         // 0..3  (32-row slab)
    const int nw   = wv & 1;          // 0..1  (64-col slab)

    const unsigned colBase = blockIdx.x * 128;
    const unsigned rowBase = blockIdx.y * 128;

    v8f acc[2][4];
    #pragma unroll
    for (int i = 0; i < 2; ++i)
        #pragma unroll
        for (int j = 0; j < 4; ++j) acc[i][j] = zero8();

    const int ar = tid & 127;         // A-stage row 0..127
    const int aq = tid >> 7;          // A-stage 16-col chunk 0..1
    const int bn = tid >> 1;          // B-stage col 0..127
    const int bq = tid & 1;           // B-stage 16-K chunk 0..1

    // One K-step (32 wide): stage A+B into LDS, then 2x4 WMMA per wave.
    auto kstep = [&](int k0, float mult, float phase) {
        __syncthreads();              // previous iteration's LDS reads done

        // ---- stage A (128 rows x 32 K), 16 f16 per thread ----
        if (FEAT) {
            const float* Af = (const float*)Ap;
            const int t0 = k0 & 511;
            const float* ap = Af + (rowBase + ar) * 512u + t0 + aq * 16;
            __builtin_prefetch(ap + 32, 0, 1);          // next K tile
            v4f x0 = *(const v4f*)ap;
            v4f x1 = *(const v4f*)(ap + 4);
            v4f x2 = *(const v4f*)(ap + 8);
            v4f x3 = *(const v4f*)(ap + 12);
            v8h h0, h1;
            #pragma unroll
            for (int i = 0; i < 4; ++i) {
                h0[i]     = (_Float16)__cosf(fmaf(mult, x0[i], phase));
                h0[4 + i] = (_Float16)__cosf(fmaf(mult, x1[i], phase));
                h1[i]     = (_Float16)__cosf(fmaf(mult, x2[i], phase));
                h1[4 + i] = (_Float16)__cosf(fmaf(mult, x3[i], phase));
            }
            *(v8h*)(lA + ar * LDA + aq * 16)     = h0;
            *(v8h*)(lA + ar * LDA + aq * 16 + 8) = h1;
        } else {
            const _Float16* Ah = (const _Float16*)Ap;
            const _Float16* ap = Ah + (rowBase + ar) * (unsigned)K + k0 + aq * 16;
            __builtin_prefetch(ap + 32, 0, 1);
            *(v8h*)(lA + ar * LDA + aq * 16)     = *(const v8h*)ap;
            *(v8h*)(lA + ar * LDA + aq * 16 + 8) = *(const v8h*)(ap + 8);
        }

        // ---- stage B (128 cols x 32 K) from N-major Bw, b128 stores ----
        {
            const _Float16* bp = Bw + (colBase + bn) * (unsigned)K + k0 + bq * 16;
            __builtin_prefetch(bp + 32, 0, 1);
            *(v8h*)(lB + bn * LDB + bq * 16)     = *(const v8h*)bp;
            *(v8h*)(lB + bn * LDB + bq * 16 + 8) = *(const v8h*)(bp + 8);
        }
        __syncthreads();

        // ---- fragments ----
        // A 16x32 f16 layout: lane half h holds K = h*8+j (j<8), 16+h*8+(j-8)
        v16h afrag[2], bfrag[4];
        #pragma unroll
        for (int ms = 0; ms < 2; ++ms) {
            const _Float16* arow = lA + (mw * 32 + ms * 16 + l16) * LDA;
            V16U u;
            u.h[0] = *(const v8h*)(arow + half * 8);
            u.h[1] = *(const v8h*)(arow + 16 + half * 8);
            afrag[ms] = u.v;
        }
        // B 32x16 f16 layout: lane half h holds K = h*8+j (j<8), 16+h*8+(j-8)
        #pragma unroll
        for (int ns = 0; ns < 4; ++ns) {
            const _Float16* brow = lB + (nw * 64 + ns * 16 + l16) * LDB;
            V16U u;
            u.h[0] = *(const v8h*)(brow + half * 8);
            u.h[1] = *(const v8h*)(brow + 16 + half * 8);
            bfrag[ns] = u.v;
        }

        #pragma unroll
        for (int ms = 0; ms < 2; ++ms)
            #pragma unroll
            for (int ns = 0; ns < 4; ++ns)
                acc[ms][ns] = __builtin_amdgcn_wmma_f32_16x16x32_f16(
                    false, afrag[ms], false, bfrag[ns],
                    (short)0, acc[ms][ns], false, false);
    };

    if (FEAT) {
        // K = 6*512: hoist g-dependent constants; fully branchless trig.
        #pragma unroll 1
        for (int g = 0; g < 6; ++g) {
            const float mult  = (g < 3) ? (float)(g + 1) : (float)(g - 2);
            const float phase = (g < 3) ? 0.0f : -1.57079632679f;  // sin = cos(x-pi/2)
            #pragma unroll 1
            for (int kt = 0; kt < 512; kt += 32)
                kstep(g * 512 + kt, mult, phase);
        }
    } else {
        #pragma unroll 1
        for (int k0 = 0; k0 < K; k0 += 32)
            kstep(k0, 0.0f, 0.0f);
    }

    // ---- epilogue (32-bit indexing; max index 16.7M) ----
    // C/D 16x16 f32 layout: lane, vgpr r -> M = r + 8*(lane>>4), N = lane&15
    const unsigned rowW = rowBase + mw * 32;
    const unsigned colW = colBase + nw * 64;
    #pragma unroll
    for (int ms = 0; ms < 2; ++ms) {
        #pragma unroll
        for (int ns = 0; ns < 4; ++ns) {
            const unsigned gcol = colW + ns * 16 + l16;
            const float bv = bias[gcol];
            #pragma unroll
            for (int r = 0; r < 8; ++r) {
                const unsigned grow = rowW + ms * 16 + half * 8 + r;
                const float val = acc[ms][ns][r] + bv;
                if (OUT_F16) {
                    ((_Float16*)outp)[grow * (unsigned)N + gcol] = (_Float16)val;
                } else {
                    unsigned idx;
                    if (TRANS_OUT)  // row n=(b,c), col t -> x2[b][t][c]
                        idx = ((grow >> 9) << 18) + (gcol << 9) + (grow & 511);
                    else
                        idx = grow * (unsigned)CC + gcol;
                    ((float*)outp)[idx] = resid[idx] + val;
                }
            }
        }
    }
}

// ---------------------------------------------------------------------------
extern "C" void kernel_launch(void* const* d_in, const int* in_sizes, int n_in,
                              void* d_out, int out_size, void* d_ws, size_t ws_size,
                              hipStream_t stream) {
    (void)in_sizes; (void)n_in; (void)out_size; (void)ws_size;

    const float* x         = (const float*)d_in[0];
    const float* ln1_w     = (const float*)d_in[1];
    const float* ln1_b     = (const float*)d_in[2];
    const float* tok_coef  = (const float*)d_in[3];
    const float* tok_kbias = (const float*)d_in[4];
    const float* tok_lw    = (const float*)d_in[5];
    const float* tok_lb    = (const float*)d_in[6];
    const float* ln2_w     = (const float*)d_in[7];
    const float* ln2_b     = (const float*)d_in[8];
    const float* ch_coef   = (const float*)d_in[9];
    const float* ch_kbias  = (const float*)d_in[10];
    const float* ch_lw     = (const float*)d_in[11];
    const float* ch_lb     = (const float*)d_in[12];
    float* out = (float*)d_out;

    char* ws = (char*)d_ws;
    size_t off = 0;
    float*     ylnT    = (float*)(ws + off);     off += (size_t)BB*CC*TT*4;      // 32 MB (reused as LN2 out)
    float*     x2      = (float*)(ws + off);     off += (size_t)BB*TT*CC*4;      // 32 MB
    _Float16*  tok_y   = (_Float16*)(ws + off);  off += (size_t)NROWS*TD*2;      // 8 MB
    _Float16*  ch_y    = (_Float16*)(ws + off);  off += (size_t)NROWS*2*CC*2;    // 32 MB
    _Float16*  W_tok   = (_Float16*)(ws + off);  off += (size_t)TD*3072*2;       // 1.5 MB  (TD x 3072)
    _Float16*  W_ch    = (_Float16*)(ws + off);  off += (size_t)2*CC*3072*2;     // 6 MB    (2C x 3072)
    _Float16*  tok_lwh = (_Float16*)(ws + off);  off += (size_t)TT*TD*2;         // 0.25 MB (T x TD, N-major)
    _Float16*  ch_lwh  = (_Float16*)(ws + off);  off += (size_t)CC*2*CC*2;       // 1 MB    (C x 2C, N-major)

    // 1) LN1 + transpose -> ylnT (B, C, T)
    ln_kernel<<<NROWS, 256, 0, stream>>>(x, ln1_w, ln1_b, ylnT, 1);

    // 2) weight repacks (N-major)
    prep_w_kernel<<<dim3(3072 / 256, TD),     256, 0, stream>>>(tok_coef, W_tok, TD);
    prep_w_kernel<<<dim3(3072 / 256, 2 * CC), 256, 0, stream>>>(ch_coef,  W_ch,  2 * CC);
    cvt_f16_kernel<<<(TT * TD) / 256,     256, 0, stream>>>(tok_lw, tok_lwh);
    cvt_f16_kernel<<<(CC * 2 * CC) / 256, 256, 0, stream>>>(ch_lw,  ch_lwh);

    // 3) token KAN: (16384 x 3072) x (3072 x 256) -> tok_y f16
    wmma_gemm_kernel<true, true, false><<<dim3(TD / 128, NROWS / 128), 256, 0, stream>>>(
        ylnT, W_tok, tok_kbias, nullptr, tok_y, 3072, TD);

    // 4) token linear + residual + un-transpose: x2 = x + tok_y @ lw^T
    wmma_gemm_kernel<false, false, true><<<dim3(TT / 128, NROWS / 128), 256, 0, stream>>>(
        tok_y, tok_lwh, tok_lb, x, x2, TD, TT);

    // 5) LN2 -> z (reuses ylnT buffer), row-major (B*T, C)
    ln_kernel<<<NROWS, 256, 0, stream>>>(x2, ln2_w, ln2_b, ylnT, 0);

    // 6) channel KAN: (16384 x 3072) x (3072 x 1024) -> ch_y f16
    wmma_gemm_kernel<true, true, false><<<dim3((2 * CC) / 128, NROWS / 128), 256, 0, stream>>>(
        ylnT, W_ch, ch_kbias, nullptr, ch_y, 3072, 2 * CC);

    // 7) channel linear + residual -> d_out
    wmma_gemm_kernel<false, false, false><<<dim3(CC / 128, NROWS / 128), 256, 0, stream>>>(
        ch_y, ch_lwh, ch_lb, x2, out, 2 * CC, CC);
}
